// CATBlock_10479720202411
// MI455X (gfx1250) — matrix operand order repaired and checked by reading, not compiled, not run on hardware
//
#include <hip/hip_runtime.h>
#include <cstdint>
#include <cstddef>

// ---------------------------------------------------------------------------
// CAT / Swin window-attention block for gfx1250 (MI455X), bf16 WMMA pipeline.
// ---------------------------------------------------------------------------

typedef __attribute__((ext_vector_type(16))) __bf16 v16bf;
typedef __attribute__((ext_vector_type(8)))  float  v8f;
typedef __bf16 bf16;

#define C_DIM   192
#define HEADS   6
#define HD      32
#define NTOK    49            // tokens per window
#define NWIN    2048          // 32 * 8 * 8
#define MROWS   100352        // B * H * W
#define SCALE_F 0.17677669529663689f   // 32^-0.5

__device__ __forceinline__ bf16  f2bf(float f) { return (bf16)f; }
__device__ __forceinline__ float bf2f(bf16 h)  { return (float)h; }

// ---- WMMA fragment loaders (contiguous k = 16*hi + s order in A and B, wave32) --------------------
// A matrix 16x32 bf16, row-major source with leading dim ld.
__device__ __forceinline__ v16bf load_a_frag(const bf16* base, int ld,
                                             int row0, int k0, int lane) {
  const int l  = lane & 15;
  const int hi = lane >> 4;
  const bf16* p = base + (size_t)(row0 + l) * ld + k0;
  v16bf a;
#pragma unroll
  for (int v = 0; v < 8; ++v) {
    const int kp = 2 * v + (hi ? 16 : 0);
    a[2 * v]     = p[kp];
    a[2 * v + 1] = p[kp + 1];
  }
  return a;
}

// B matrix 32x16 bf16 from row-major K x N source (B[k][n]), leading dim ld.
__device__ __forceinline__ v16bf load_b_frag(const bf16* base, int ld,
                                             int k0, int col0, int lane) {
  const int l  = lane & 15;
  const int hi = lane >> 4;
  v16bf b;
#pragma unroll
  for (int v = 0; v < 8; ++v) {
    const int k = k0 + (hi ? 16 : 0) + 2 * v;
    b[2 * v]     = base[(size_t)k       * ld + col0 + l];
    b[2 * v + 1] = base[(size_t)(k + 1) * ld + col0 + l];
  }
  return b;
}

// B matrix 32x16 where logical B[k][n] = src[n][k] (i.e. N x K row-major src).
__device__ __forceinline__ v16bf load_bT_frag(const bf16* base, int ld,
                                              int k0, int col0, int lane) {
  const int l  = lane & 15;
  const int hi = lane >> 4;
  const bf16* p = base + (size_t)(col0 + l) * ld + k0 + (hi ? 16 : 0);
  v16bf b;
#pragma unroll
  for (int v = 0; v < 8; ++v) {
    b[2 * v]     = p[2 * v];
    b[2 * v + 1] = p[2 * v + 1];
  }
  return b;
}

__device__ __forceinline__ v8f wmma_bf16(v16bf a, v16bf b, v8f c) {
  return __builtin_amdgcn_wmma_f32_16x16x32_bf16(false, a, false, b,
                                                 (short)0, c, false, false);
}

// ---------------------------------------------------------------------------
// fp32 -> bf16 weight conversion
// ---------------------------------------------------------------------------
__global__ __launch_bounds__(256) void f2bf_kernel(const float* __restrict__ in,
                                                   bf16* __restrict__ out, int n) {
  const int i = blockIdx.x * 256 + threadIdx.x;
  if (i < n) out[i] = f2bf(in[i]);
}

// ---------------------------------------------------------------------------
// Precompute attention bias tiles: biasg[h][m][c] (64x64 per head, padded).
// c >= 49  -> -1e30 (masks padded key columns through softmax)
// m >= 49  -> 0     (padded query rows, values never read back)
// ---------------------------------------------------------------------------
__global__ __launch_bounds__(256) void bias_kernel(const float* __restrict__ rpe,
                                                   float* __restrict__ biasg) {
  const int i = blockIdx.x * 256 + threadIdx.x;   // < 6*64*64
  const int h = i >> 12;
  const int m = (i >> 6) & 63;
  const int c = i & 63;
  float v;
  if (c >= NTOK)      v = -1e30f;
  else if (m >= NTOK) v = 0.f;
  else {
    const int dy = m / 7 - c / 7 + 6;
    const int dx = m % 7 - c % 7 + 6;
    v = rpe[(dy * 13 + dx) * HEADS + h];
  }
  biasg[i] = v;
}

// ---------------------------------------------------------------------------
// LayerNorm over C=192, one wave32 per row, 8 rows per block.
// ---------------------------------------------------------------------------
__global__ __launch_bounds__(256) void ln_kernel(const float* __restrict__ x,
                                                 const float* __restrict__ g,
                                                 const float* __restrict__ b,
                                                 bf16* __restrict__ out) {
  const int wv   = threadIdx.x >> 5;
  const int lane = threadIdx.x & 31;
  const int row  = blockIdx.x * 8 + wv;
  const float* xr = x + (size_t)row * C_DIM;
  float v[6], s = 0.f, sq = 0.f;
#pragma unroll
  for (int j = 0; j < 6; ++j) {
    v[j] = xr[lane + 32 * j];
    s += v[j];
    sq += v[j] * v[j];
  }
#pragma unroll
  for (int o = 16; o > 0; o >>= 1) {
    s  += __shfl_xor(s,  o, 32);
    sq += __shfl_xor(sq, o, 32);
  }
  const float mu  = s * (1.f / C_DIM);
  const float var = sq * (1.f / C_DIM) - mu * mu;
  const float rs  = rsqrtf(var + 1e-5f);
  bf16* orow = out + (size_t)row * C_DIM;
#pragma unroll
  for (int j = 0; j < 6; ++j) {
    const int c = lane + 32 * j;
    orow[c] = f2bf((v[j] - mu) * rs * g[c] + b[c]);
  }
}

// ---------------------------------------------------------------------------
// Tiled bf16 WMMA GEMM: out[M,N] = A[M,K] @ W[K,N] + bias (+gelu) (+resid)
// Block tile 256x64, BK=32, 8 waves (4Mx2N) each computing 64x32
// (4x2 WMMA tiles -> 8 v_wmma per K-step, 6 fragment loads).
// Global loads for tile kt+1 are issued before the WMMA work on tile kt
// so the memory latency overlaps the matrix pipe.
// M%256==0, N%64==0, K%32==0 (true for all calls here).
// ---------------------------------------------------------------------------
template <bool OUT_BF16, bool GELU, bool RESID>
__global__ __launch_bounds__(256) void gemm_bf16_kernel(
    const bf16* __restrict__ A, const bf16* __restrict__ W,
    const float* __restrict__ bias, const float* __restrict__ resid,
    void* __restrict__ out, int M, int N, int K) {
  __shared__ bf16 As[256][40];   // 256x32 tile, rows padded to 80B
  __shared__ bf16 Bs[32][72];    // 32x64 tile, rows padded to 144B

  const int tid   = threadIdx.x;
  const int lane  = tid & 31;
  const int wv    = tid >> 5;
  const int waveM = wv & 3;      // 0..3 -> row offset 0,64,128,192
  const int waveN = wv >> 2;     // 0..1 -> col offset 0,32
  const int l     = lane & 15;
  const int hi    = lane >> 4;

  const v8f zero = {0.f, 0.f, 0.f, 0.f, 0.f, 0.f, 0.f, 0.f};
  v8f acc[4][2];
#pragma unroll
  for (int i = 0; i < 4; ++i) { acc[i][0] = zero; acc[i][1] = zero; }

  // staging: thread tid owns A-tile row tid (32 bf16 = 4 x b128)
  const size_t aRow = (size_t)blockIdx.y * 256 + tid;
  const int br = tid >> 3;             // B tile row 0..31
  const int bc = (tid & 7) * 8;        // B tile col
  const size_t bCol = (size_t)blockIdx.x * 64 + bc;

  const int ksteps = K >> 5;
  uint4 aReg[4], bReg;

  // prologue: fetch tile 0
  {
    const uint4* asrc = reinterpret_cast<const uint4*>(A + aRow * K);
#pragma unroll
    for (int j = 0; j < 4; ++j) aReg[j] = asrc[j];
    bReg = *reinterpret_cast<const uint4*>(W + (size_t)br * N + bCol);
  }
#pragma unroll
  for (int j = 0; j < 4; ++j)
    reinterpret_cast<uint4*>(&As[tid][0])[j] = aReg[j];
  *reinterpret_cast<uint4*>(&Bs[br][bc]) = bReg;
  __syncthreads();

  for (int kt = 0; kt < ksteps; ++kt) {
    // issue next tile's global loads first (overlap with WMMA below)
    if (kt + 1 < ksteps) {
      const uint4* asrc =
          reinterpret_cast<const uint4*>(A + aRow * K + (kt + 1) * 32);
#pragma unroll
      for (int j = 0; j < 4; ++j) aReg[j] = asrc[j];
      bReg = *reinterpret_cast<const uint4*>(
          W + (size_t)((kt + 1) * 32 + br) * N + bCol);
    }

    const v16bf b0 = load_b_frag(&Bs[0][0], 72, 0, waveN * 32,      lane);
    const v16bf b1 = load_b_frag(&Bs[0][0], 72, 0, waveN * 32 + 16, lane);
#pragma unroll
    for (int i = 0; i < 4; ++i) {
      const v16bf a = load_a_frag(&As[0][0], 40, waveM * 64 + i * 16, 0, lane);
      acc[i][0] = wmma_bf16(a, b0, acc[i][0]);
      acc[i][1] = wmma_bf16(a, b1, acc[i][1]);
    }
    __syncthreads();

    if (kt + 1 < ksteps) {
#pragma unroll
      for (int j = 0; j < 4; ++j)
        reinterpret_cast<uint4*>(&As[tid][0])[j] = aReg[j];
      *reinterpret_cast<uint4*>(&Bs[br][bc]) = bReg;
      __syncthreads();
    }
  }

  // epilogue: bias (+gelu) (+residual), store bf16 or fp32
#pragma unroll
  for (int i = 0; i < 4; ++i)
#pragma unroll
    for (int j = 0; j < 2; ++j)
#pragma unroll
      for (int r = 0; r < 8; ++r) {
        const int m = waveM * 64 + i * 16 + r + hi * 8;
        const int n = waveN * 32 + j * 16 + l;
        const size_t grow = (size_t)blockIdx.y * 256 + m;
        const int    gcol = blockIdx.x * 64 + n;
        float v = acc[i][j][r] + bias[gcol];
        if (GELU) v = 0.5f * v * (1.f + erff(v * 0.70710678118654752f));
        if (RESID) v += resid[grow * N + gcol];
        if (OUT_BF16)
          reinterpret_cast<bf16*>(out)[grow * N + gcol] = f2bf(v);
        else
          reinterpret_cast<float*>(out)[grow * N + gcol] = v;
      }
}

// ---------------------------------------------------------------------------
// Window attention: one block (4 waves) per (window, head).
// 49 tokens padded to 64; S = (q*scale) k^T + precomputed bias; softmax;
// O = S v.  Scatters O back to x-row order so proj GEMM runs in row order.
// ---------------------------------------------------------------------------
__global__ __launch_bounds__(128) void attn_kernel(const bf16* __restrict__ qkv,
                                                   const float* __restrict__ biasg,
                                                   bf16* __restrict__ out) {
  const int head = blockIdx.x % HEADS;
  const int w    = blockIdx.x / HEADS;
  const int bb   = w >> 6;
  const int wy   = (w >> 3) & 7;
  const int wx   = w & 7;

  const int tid  = threadIdx.x;
  const int lane = tid & 31;
  const int wv   = tid >> 5;          // 0..3
  const int l    = lane & 15;
  const int hi   = lane >> 4;

  __shared__ bf16 qs[64][32];
  __shared__ bf16 ksm[64][32];
  __shared__ bf16 vsm[64][32];
  __shared__ float Sf[64][65];
  __shared__ bf16  Sb[64][72];

  const float* biasT = biasg + (size_t)head * 64 * 64;

  // gather q,k,v for the window's 49 tokens (pad rows 49..63 with zeros)
  for (int i = tid; i < 64 * 32; i += 128) {
    const int n = i >> 5, d = i & 31;
    bf16 qv = f2bf(0.f), kv = qv, vv = qv;
    if (n < NTOK) {
      const int py = n / 7, px = n % 7;
      const size_t row  = (size_t)(bb * 56 + wy * 7 + py) * 56 + (wx * 7 + px);
      const size_t base = row * 576 + head * HD + d;
      qv = f2bf(bf2f(qkv[base]) * SCALE_F);
      kv = qkv[base + 192];
      vv = qkv[base + 384];
    }
    qs[n][d] = qv;
    ksm[n][d] = kv;
    vsm[n][d] = vv;
  }
  __syncthreads();

  // ---- S = q k^T : wave wv owns rows [16wv, 16wv+16), 4 col tiles, K=32
  const int rowBase = wv * 16;
  const v8f zero = {0.f, 0.f, 0.f, 0.f, 0.f, 0.f, 0.f, 0.f};
  v8f sacc[4] = {zero, zero, zero, zero};
  const v16bf aq = load_a_frag(&qs[0][0], 32, rowBase, 0, lane);
#pragma unroll
  for (int nt = 0; nt < 4; ++nt) {
    const v16bf bk = load_bT_frag(&ksm[0][0], 32, 0, nt * 16, lane);
    sacc[nt] = wmma_bf16(aq, bk, sacc[nt]);
  }

  // + precomputed bias/mask (branch-free, coalesced), spill to LDS fp32
#pragma unroll
  for (int nt = 0; nt < 4; ++nt)
#pragma unroll
    for (int r = 0; r < 8; ++r) {
      const int m = rowBase + r + hi * 8;     // query token
      const int c = nt * 16 + l;              // key token
      Sf[m][c] = sacc[nt][r] + biasT[m * 64 + c];
    }
  __syncthreads();

  // softmax: one thread per row
  if (tid < 64) {
    float mx = -1e30f;
    for (int c = 0; c < NTOK; ++c) mx = fmaxf(mx, Sf[tid][c]);
    float sum = 0.f;
    for (int c = 0; c < NTOK; ++c) {
      const float e = __expf(Sf[tid][c] - mx);
      sum += e;
      Sf[tid][c] = e;
    }
    const float inv = 1.f / sum;
    for (int c = 0; c < 64; ++c)
      Sb[tid][c] = (c < NTOK) ? f2bf(Sf[tid][c] * inv) : f2bf(0.f);
  }
  __syncthreads();

  // ---- O = Ssoft @ v : rows [16wv,+16), 2 col tiles (HD=32), K=64 (2 steps)
  v8f oacc[2] = {zero, zero};
#pragma unroll
  for (int ks = 0; ks < 2; ++ks) {
    const v16bf as = load_a_frag(&Sb[0][0], 72, rowBase, ks * 32, lane);
#pragma unroll
    for (int dt = 0; dt < 2; ++dt) {
      const v16bf bv = load_b_frag(&vsm[0][0], 32, ks * 32, dt * 16, lane);
      oacc[dt] = wmma_bf16(as, bv, oacc[dt]);
    }
  }

  // scatter to x-row order bf16
#pragma unroll
  for (int dt = 0; dt < 2; ++dt)
#pragma unroll
    for (int r = 0; r < 8; ++r) {
      const int m = rowBase + r + hi * 8;
      if (m < NTOK) {
        const int d  = dt * 16 + l;
        const int py = m / 7, px = m % 7;
        const size_t row = (size_t)(bb * 56 + wy * 7 + py) * 56 + (wx * 7 + px);
        out[row * C_DIM + head * HD + d] = f2bf(oacc[dt][r]);
      }
    }
}

// ---------------------------------------------------------------------------
// Host-side orchestration
// ---------------------------------------------------------------------------
extern "C" void kernel_launch(void* const* d_in, const int* in_sizes, int n_in,
                              void* d_out, int out_size, void* d_ws, size_t ws_size,
                              hipStream_t stream) {
  (void)in_sizes; (void)n_in; (void)out_size; (void)ws_size;
  const float* x       = (const float*)d_in[0];
  const float* norm1_g = (const float*)d_in[1];
  const float* norm1_b = (const float*)d_in[2];
  const float* qkv_w   = (const float*)d_in[3];
  const float* qkv_b   = (const float*)d_in[4];
  const float* rpe     = (const float*)d_in[5];
  const float* proj_w  = (const float*)d_in[6];
  const float* proj_b  = (const float*)d_in[7];
  const float* norm2_g = (const float*)d_in[8];
  const float* norm2_b = (const float*)d_in[9];
  const float* fc1_w   = (const float*)d_in[10];
  const float* fc1_b   = (const float*)d_in[11];
  const float* fc2_w   = (const float*)d_in[12];
  const float* fc2_b   = (const float*)d_in[13];
  float* out = (float*)d_out;

  const size_t M = MROWS;
  char* ws = (char*)d_ws;
  // workspace layout (bytes); region 1 is reused (qkv then fc1 activations)
  const size_t off_hb   = 0;                                   // M*192 bf16
  const size_t off_big  = off_hb  + M * 192 * 2;               // max(M*576, M*768) bf16
  const size_t off_attn = off_big + M * 768 * 2;               // M*192 bf16
  const size_t off_w    = off_attn + M * 192 * 2;              // bf16 weights
  const size_t off_bias = off_w + 2 * (192 * 576 + 192 * 192 + 192 * 768 + 768 * 192);
  bf16* hb     = (bf16*)(ws + off_hb);     // LN1 out, later LN2 out
  bf16* qkvb   = (bf16*)(ws + off_big);    // QKV activations
  bf16* m1b    = (bf16*)(ws + off_big);    // gelu(fc1) activations (reuse)
  bf16* attn_o = (bf16*)(ws + off_attn);
  bf16* qkvwb  = (bf16*)(ws + off_w);
  bf16* projwb = qkvwb  + 192 * 576;
  bf16* fc1wb  = projwb + 192 * 192;
  bf16* fc2wb  = fc1wb  + 192 * 768;
  float* biasg = (float*)(ws + off_bias);  // 6*64*64 fp32

  // 0) weights fp32 -> bf16; attention bias tiles
  f2bf_kernel<<<(192 * 576 + 255) / 256, 256, 0, stream>>>(qkv_w,  qkvwb,  192 * 576);
  f2bf_kernel<<<(192 * 192 + 255) / 256, 256, 0, stream>>>(proj_w, projwb, 192 * 192);
  f2bf_kernel<<<(192 * 768 + 255) / 256, 256, 0, stream>>>(fc1_w,  fc1wb,  192 * 768);
  f2bf_kernel<<<(768 * 192 + 255) / 256, 256, 0, stream>>>(fc2_w,  fc2wb,  768 * 192);
  bias_kernel<<<(HEADS * 64 * 64) / 256, 256, 0, stream>>>(rpe, biasg);

  // 1) LN1
  ln_kernel<<<M / 8, 256, 0, stream>>>(x, norm1_g, norm1_b, hb);

  // 2) QKV GEMM  [M,192] @ [192,576]
  gemm_bf16_kernel<true, false, false><<<dim3(576 / 64, M / 256), 256, 0, stream>>>(
      hb, qkvwb, qkv_b, nullptr, qkvb, (int)M, 576, 192);

  // 3) window attention, one block per (window, head)
  attn_kernel<<<NWIN * HEADS, 128, 0, stream>>>(qkvb, biasg, attn_o);

  // 4) proj GEMM + bias + shortcut residual -> fp32 in d_out
  gemm_bf16_kernel<false, false, true><<<dim3(192 / 64, M / 256), 256, 0, stream>>>(
      attn_o, projwb, proj_b, x, out, (int)M, 192, 192);

  // 5) LN2 (reads d_out)
  ln_kernel<<<M / 8, 256, 0, stream>>>(out, norm2_g, norm2_b, hb);

  // 6) fc1 GEMM + GELU  [M,192] @ [192,768]
  gemm_bf16_kernel<true, true, false><<<dim3(768 / 64, M / 256), 256, 0, stream>>>(
      hb, fc1wb, fc1_b, nullptr, m1b, (int)M, 768, 192);

  // 7) fc2 GEMM + bias + residual, in-place on d_out
  gemm_bf16_kernel<false, false, true><<<dim3(192 / 64, M / 256), 256, 0, stream>>>(
      m1b, fc2wb, fc2_b, out, out, (int)M, 192, 768);
}
